// GCNWithEdge_52312701665557
// MI455X (gfx1250) — matrix operand order repaired and checked by reading, not compiled, not run on hardware
//
#include <hip/hip_runtime.h>

typedef __attribute__((ext_vector_type(16))) _Float16 v16h;
typedef __attribute__((ext_vector_type(8)))  _Float16 v8h;
typedef __attribute__((ext_vector_type(2)))  _Float16 v2h;
typedef __attribute__((ext_vector_type(8)))  float    v8f;

// ---------------------------------------------------------------- utilities
__global__ void k_fill1(float* __restrict__ p, int n) {
  int i = blockIdx.x * blockDim.x + threadIdx.x;
  if (i < n) p[i] = 1.0f;
}

__global__ void k_zero(float* __restrict__ p, long long n) {
  long long i = (long long)blockIdx.x * blockDim.x + threadIdx.x;
  if (i < n) p[i] = 0.0f;
}

__global__ void k_deg_count(const int* __restrict__ dstn, float* __restrict__ deg, int E) {
  int e = blockIdx.x * blockDim.x + threadIdx.x;
  if (e < E)
    __hip_atomic_fetch_add(&deg[dstn[e]], 1.0f, __ATOMIC_RELAXED, __HIP_MEMORY_SCOPE_AGENT);
}

__global__ void k_rsqrt(float* __restrict__ p, int n) {
  int i = blockIdx.x * blockDim.x + threadIdx.x;
  if (i < n) p[i] = rsqrtf(p[i]);
}

// f32 -> f16, 2 elements per thread (v_cvt_pk-friendly). n must be even.
__global__ void k_f32_to_f16(const float* __restrict__ in, _Float16* __restrict__ out,
                             long long n2) {
  long long i = (long long)blockIdx.x * blockDim.x + threadIdx.x;
  if (i >= n2) return;
  float2 v = ((const float2*)in)[i];
  v2h h; h[0] = (_Float16)v.x; h[1] = (_Float16)v.y;
  ((v2h*)out)[i] = h;
}

// W (K x F, f32) -> Wt (F x K, f16)
__global__ void k_w_to_f16t(const float* __restrict__ W, _Float16* __restrict__ Wt,
                            int K, int F) {
  int i = blockIdx.x * blockDim.x + threadIdx.x;
  if (i >= K * F) return;
  int k = i / F, f = i - k * F;
  Wt[(size_t)f * K + k] = (_Float16)W[i];
}

// ------------------------------------------------- WMMA GEMM (f16 inputs, f32 out)
// C[M,F] = A16[M,K] * Bt16[F,K]^T. One wave per 16x16 tile, K fully unrolled,
// v_wmma_f32_16x16x32_f16. All control flow around the WMMAs is wave-uniform.
template <int K, int F>
__global__ void __launch_bounds__(256)
k_gemm_wmma(const _Float16* __restrict__ A, const _Float16* __restrict__ Bt,
            float* __restrict__ C, int M) {
  int gw   = (blockIdx.x * blockDim.x + threadIdx.x) >> 5;
  int lane = threadIdx.x & 31;
  constexpr int tilesPerRow = F >> 4;
  int nTiles = ((M + 15) >> 4) * tilesPerRow;
  if (gw >= nTiles) return;                     // wave-uniform
  int tRow = gw / tilesPerRow;
  int tCol = gw - tRow * tilesPerRow;
  int l16 = lane & 15;
  int hi  = lane >> 4;
  int m  = tRow * 16 + l16;
  int mC = m < M ? m : (M - 1);                 // clamp (loads only)
  int n  = tCol * 16 + l16;

  // A fragment (16-bit 16x32 layout): elems 0..7 -> K=k0+hi*8+e (contig 8 halfs),
  // elems 8..15 -> K=k0+16+hi*8+e (contig 8 halfs).
  const _Float16* aBase = A + (size_t)mC * K + hi * 8;
  // B fragment: same K mapping as A (elems 0..7 -> k0+hi*8+e, 8..15 -> k0+16+hi*8+e).
  const _Float16* bBase = Bt + (size_t)n * K + hi * 8;

  v8f acc = {};
#pragma unroll
  for (int k0 = 0; k0 < K; k0 += 32) {
    v8h alo = *(const v8h*)(aBase + k0);
    v8h ahi = *(const v8h*)(aBase + k0 + 16);
    v16h av = __builtin_shufflevector(alo, ahi, 0, 1, 2, 3, 4, 5, 6, 7,
                                      8, 9, 10, 11, 12, 13, 14, 15);
    v8h blo = *(const v8h*)(bBase + k0);
    v8h bhi = *(const v8h*)(bBase + k0 + 16);
    v16h bv = __builtin_shufflevector(blo, bhi, 0, 1, 2, 3, 4, 5, 6, 7,
                                      8, 9, 10, 11, 12, 13, 14, 15);
    acc = __builtin_amdgcn_wmma_f32_16x16x32_f16(
        /*neg_a=*/false, av, /*neg_b=*/false, bv,
        /*c_mod=*/(short)0, acc, /*reuse_a=*/false, /*reuse_b=*/false);
  }

  // C/D layout: VGPR v -> row = tRow*16 + hi*8 + v, col = n
  int rowBase = tRow * 16 + hi * 8;
  float* cp = C + (size_t)rowBase * F + n;
  if (tRow * 16 + 16 <= M) {                    // wave-uniform full-tile path
#pragma unroll
    for (int v = 0; v < 8; ++v) cp[(size_t)v * F] = acc[v];
  } else {
#pragma unroll
    for (int v = 0; v < 8; ++v)
      if (rowBase + v < M) cp[(size_t)v * F] = acc[v];
  }
}

// ---------------------------------------- scatter: agg[dst] += h[src] * dinv[src]*dinv[dst]
__global__ void k_scatter(const float* __restrict__ hlin, const int* __restrict__ srcn,
                          const int* __restrict__ dstn, const float* __restrict__ dinv,
                          float* __restrict__ agg, int E, int F, int fshift) {
  long long tid = (long long)blockIdx.x * blockDim.x + threadIdx.x;
  long long total = (long long)E << fshift;       // fshift = log2(F/4)
  if (tid >= total) return;
  int e  = (int)(tid >> fshift);
  int f4 = (int)(tid & ((1 << fshift) - 1));
  int s = srcn[e], d = dstn[e];
  float nrm = dinv[s] * dinv[d];
  float4 hv = ((const float4*)(hlin + (size_t)s * F))[f4];
  float* p = agg + (size_t)d * F + (size_t)f4 * 4;
  __hip_atomic_fetch_add(p + 0, hv.x * nrm, __ATOMIC_RELAXED, __HIP_MEMORY_SCOPE_AGENT);
  __hip_atomic_fetch_add(p + 1, hv.y * nrm, __ATOMIC_RELAXED, __HIP_MEMORY_SCOPE_AGENT);
  __hip_atomic_fetch_add(p + 2, hv.z * nrm, __ATOMIC_RELAXED, __HIP_MEMORY_SCOPE_AGENT);
  __hip_atomic_fetch_add(p + 3, hv.w * nrm, __ATOMIC_RELAXED, __HIP_MEMORY_SCOPE_AGENT);
}

// ---- layer-1 combine: h1 = relu(agg + hlin*dinv^2 + b1), written directly as f16 (F=256)
__global__ void k_combine1(const float* __restrict__ agg, const float* __restrict__ hlin,
                           const float* __restrict__ dinv, const float* __restrict__ bias,
                           _Float16* __restrict__ h16, int N) {
  long long tid = (long long)blockIdx.x * blockDim.x + threadIdx.x;
  long long total = (long long)N << 8;
  if (tid >= total) return;
  int n = (int)(tid >> 8);
  int f = (int)(tid & 255);
  float di = dinv[n];
  float v = agg[tid] + hlin[tid] * di * di + bias[f];
  h16[tid] = (_Float16)fmaxf(v, 0.0f);
}

// ---- layer-2 combine: h2 = agg + hlin*dinv^2 + b2, f32 in place on agg (F=128)
__global__ void k_combine2(float* __restrict__ agg, const float* __restrict__ hlin,
                           const float* __restrict__ dinv, const float* __restrict__ bias,
                           int N) {
  long long tid = (long long)blockIdx.x * blockDim.x + threadIdx.x;
  long long total = (long long)N << 7;
  if (tid >= total) return;
  int n = (int)(tid >> 7);
  int f = (int)(tid & 127);
  float di = dinv[n];
  agg[tid] = agg[tid] + hlin[tid] * di * di + bias[f];
}

// --------------------------------------------------------- final edge MLP: one wave per edge
__global__ void __launch_bounds__(256)
k_edge(const float* __restrict__ h, const int* __restrict__ srcn, const int* __restrict__ dstn,
       const int* __restrict__ ecat, const float* __restrict__ eval,
       const float* __restrict__ emb0, const float* __restrict__ emb1,
       const float* __restrict__ fW1, const float* __restrict__ fb1,
       const float* __restrict__ fW2, const float* __restrict__ fb2,
       float* __restrict__ out, int E) {
  int gw   = (blockIdx.x * blockDim.x + threadIdx.x) >> 5;
  int lane = threadIdx.x & 31;
  if (gw >= E) return;                                  // wave-uniform
  int s = srcn[gw], d = dstn[gw];

  float4 a = ((const float4*)(h + (size_t)s * 128))[lane];
  float4 b = ((const float4*)(h + (size_t)d * 128))[lane];
  float v0 = a.x - b.x, v1 = a.y - b.y, v2 = a.z - b.z, v3 = a.w - b.w;
  int base = lane * 4;
  float acc0 = v0 * fW1[(base + 0) * 2 + 0] + v1 * fW1[(base + 1) * 2 + 0] +
               v2 * fW1[(base + 2) * 2 + 0] + v3 * fW1[(base + 3) * 2 + 0];
  float acc1 = v0 * fW1[(base + 0) * 2 + 1] + v1 * fW1[(base + 1) * 2 + 1] +
               v2 * fW1[(base + 2) * 2 + 1] + v3 * fW1[(base + 3) * 2 + 1];

  int c0 = ecat[gw * 2 + 0], c1 = ecat[gw * 2 + 1];
  float e0 = emb0[c0 * 32 + lane];
  acc0 += e0 * fW1[(130 + lane) * 2 + 0];
  acc1 += e0 * fW1[(130 + lane) * 2 + 1];
  float e1 = emb1[c1 * 32 + lane];
  acc0 += e1 * fW1[(162 + lane) * 2 + 0];
  acc1 += e1 * fW1[(162 + lane) * 2 + 1];
  if (lane < 2) {
    float ev = eval[gw * 2 + lane];
    acc0 += ev * fW1[(128 + lane) * 2 + 0];
    acc1 += ev * fW1[(128 + lane) * 2 + 1];
  }
#pragma unroll
  for (int off = 16; off > 0; off >>= 1) {
    acc0 += __shfl_down(acc0, off);
    acc1 += __shfl_down(acc1, off);
  }
  if (lane == 0) {
    float z0 = fmaxf(acc0 + fb1[0], 0.0f);
    float z1 = fmaxf(acc1 + fb1[1], 0.0f);
    out[(size_t)gw * 2 + 0] = z0 * fW2[0] + z1 * fW2[2] + fb2[0];
    out[(size_t)gw * 2 + 1] = z0 * fW2[1] + z1 * fW2[3] + fb2[1];
  }
}

// =============================================================================
extern "C" void kernel_launch(void* const* d_in, const int* in_sizes, int n_in,
                              void* d_out, int out_size, void* d_ws, size_t ws_size,
                              hipStream_t stream) {
  const float* x    = (const float*)d_in[0];
  const int*   eidx = (const int*)d_in[1];
  const int*   ecat = (const int*)d_in[2];
  const float* eval = (const float*)d_in[3];
  const float* W1   = (const float*)d_in[4];
  const float* b1   = (const float*)d_in[5];
  const float* W2   = (const float*)d_in[6];
  const float* b2   = (const float*)d_in[7];
  const float* emb0 = (const float*)d_in[8];
  const float* emb1 = (const float*)d_in[9];
  const float* fW1  = (const float*)d_in[10];
  const float* fb1  = (const float*)d_in[11];
  const float* fW2  = (const float*)d_in[12];
  const float* fb2  = (const float*)d_in[13];

  const int IN = 256, HID = 256, OUT = 128;
  const int N = in_sizes[0] / IN;
  const int E = in_sizes[1] / 2;
  const int* srcn = eidx;        // edge_index row 0
  const int* dstn = eidx + E;    // edge_index row 1

  // ---- workspace layout (64B-aligned regions) ----
  uintptr_t p = (uintptr_t)d_ws;
  auto take = [&](size_t bytes) { uintptr_t q = (p + 63) & ~(uintptr_t)63; p = q + bytes; return q; };
  float*    dinv = (float*)take((size_t)N * 4);
  float*    bufA = (float*)take((size_t)N * HID * 4);   // h1lin; later h2lin | agg2/h2
  float*    bufB = (float*)take((size_t)N * HID * 4);   // agg1
  _Float16* x16  = (_Float16*)take((size_t)N * IN * 2); // x in f16; reused as h1 f16
  _Float16* W1t  = (_Float16*)take((size_t)IN * HID * 2);  // HID x IN (F x K)
  _Float16* W2t  = (_Float16*)take((size_t)HID * OUT * 2); // OUT x HID (F x K)
  float*    outp = (float*)d_out;

  // ---- degrees -> dinv
  k_fill1    <<<(N + 255) / 256, 256, 0, stream>>>(dinv, N);
  k_deg_count<<<(E + 255) / 256, 256, 0, stream>>>(dstn, dinv, E);
  k_rsqrt    <<<(N + 255) / 256, 256, 0, stream>>>(dinv, N);

  // ---- one-time f16 packing (x, W1^T, W2^T)
  {
    long long n2 = (long long)N * IN / 2;
    k_f32_to_f16<<<(int)((n2 + 255) / 256), 256, 0, stream>>>(x, x16, n2);
    k_w_to_f16t<<<(IN * HID + 255) / 256, 256, 0, stream>>>(W1, W1t, IN, HID);
    k_w_to_f16t<<<(HID * OUT + 255) / 256, 256, 0, stream>>>(W2, W2t, HID, OUT);
  }

  // ---- layer 1: h1lin = x @ W1 (WMMA), aggregate, combine -> h1 (f16, reuses x16)
  {
    int tiles = ((N + 15) / 16) * (HID / 16);
    k_gemm_wmma<256, 256><<<(tiles + 7) / 8, 256, 0, stream>>>(x16, W1t, bufA, N);
  }
  {
    long long tot = (long long)N * HID;
    k_zero<<<(int)((tot + 255) / 256), 256, 0, stream>>>(bufB, tot);
    long long st = (long long)E * (HID / 4);
    k_scatter<<<(int)((st + 255) / 256), 256, 0, stream>>>(bufA, srcn, dstn, dinv, bufB, E, HID, 6);
    k_combine1<<<(int)((tot + 255) / 256), 256, 0, stream>>>(bufB, bufA, dinv, b1, x16, N);
  }

  // ---- layer 2: h2lin = h1 @ W2 (WMMA), aggregate, combine (f32)
  {
    int tiles = ((N + 15) / 16) * (OUT / 16);
    k_gemm_wmma<256, 128><<<(tiles + 7) / 8, 256, 0, stream>>>(x16, W2t, bufA, N);
  }
  float* agg2 = bufA + (size_t)N * OUT;
  {
    long long tot = (long long)N * OUT;
    k_zero<<<(int)((tot + 255) / 256), 256, 0, stream>>>(agg2, tot);
    long long st = (long long)E * (OUT / 4);
    k_scatter<<<(int)((st + 255) / 256), 256, 0, stream>>>(bufA, srcn, dstn, dinv, agg2, E, OUT, 5);
    k_combine2<<<(int)((tot + 255) / 256), 256, 0, stream>>>(agg2, bufA, dinv, b2, N);
  }

  // ---- final per-edge MLP (one wave per edge)
  k_edge<<<(E + 7) / 8, 256, 0, stream>>>(agg2, srcn, dstn, ecat, eval, emb0, emb1,
                                          fW1, fb1, fW2, fb2, outp, E);
}
